// TailGAT_SP_10866267259413
// MI455X (gfx1250) — matrix operand order repaired and checked by reading, not compiled, not run on hardware
//
#include <hip/hip_runtime.h>
#include <hip/hip_bf16.h>
#include <math.h>

// ---------------------------------------------------------------------------
// CDNA5 (gfx1250) GAT forward.
// Layer-1 relation GEMM: single pass over x/neighbor (grid.y==1, 8 col-tiles
// per wave, 16 WMMA accumulator sets), weights pre-packed into the exact WMMA
// B-fragment register image (2x global_load_b128 per fragment, L2-resident).
// Scatter/attention stages: atomic VALU kernels.
// ---------------------------------------------------------------------------

typedef __attribute__((ext_vector_type(16))) __bf16 v16bf;
typedef __attribute__((ext_vector_type(8)))  float  v8f;
typedef __attribute__((ext_vector_type(2)))  float  v2f;
typedef __attribute__((ext_vector_type(2)))  __bf16 v2bf;

#define NFEAT  512
#define NH1    8
#define NHEADS 3
#define D1     24
#define NCLASS 32

__device__ __forceinline__ float lrelu_f(float v) { return v > 0.f ? v : 0.2f * v; }
__device__ __forceinline__ float elu_f(float v)   { return v > 0.f ? v : (expf(v) - 1.f); }
__device__ __forceinline__ unsigned short f2bf(float f) {
    unsigned u = __float_as_uint(f);
    u += 0x7FFFu + ((u >> 16) & 1u);            // round-to-nearest-even
    return (unsigned short)(u >> 16);
}
// packed f32x2 -> bf16x2 (lowers to v_cvt_pk_bf16_f32 when available)
__device__ __forceinline__ unsigned pack2_bf16(float a, float b) {
    v2f f = {a, b};
    v2bf h = __builtin_convertvector(f, v2bf);
    unsigned r;
    __builtin_memcpy(&r, &h, 4);
    return r;
}

union Frag { unsigned u[8]; uint4 q[2]; v16bf v; };

// --------------------------- weight pre-packing ----------------------------
// Pack f32 row-major W[K][ldw] (taking C columns) into WMMA B-fragment layout:
// P[((ct*KT + ktIdx)*32 + lane)*8 + v] = bf16(W[k0][ct*16+n]) | bf16(W[k1][..])<<16
// with n = lane&15, half = lane>>4, k0 = ktIdx*32 + half*16 + 2v, k1 = k0+1.

__global__ void k_pack_wfrag(const float* __restrict__ W, unsigned* __restrict__ P,
                             int K, int C, int ldw) {
    int KT = K >> 5;
    int total = (C >> 4) * KT * 256;
    int t = blockIdx.x * 256 + threadIdx.x;
    if (t >= total) return;
    int v = t & 7;
    int lane = (t >> 3) & 31;
    int tile = t >> 8;
    int ktIdx = tile % KT;
    int ct = tile / KT;
    int n = lane & 15, half = lane >> 4;
    int k0 = ktIdx * 32 + half * 16 + 2 * v;
    int col = ct * 16 + n;
    unsigned lo = f2bf(W[(size_t)k0 * ldw + col]);
    unsigned hi = f2bf(W[(size_t)(k0 + 1) * ldw + col]);
    P[t] = lo | (hi << 16);
}

// gat1_W [3][512][8] -> packed fragments for a virtual [512][32] matrix
// (cols 24..31 = 0).
__global__ void k_pack_w1frag(const float* __restrict__ W, unsigned* __restrict__ P) {
    const int KT = NFEAT >> 5;               // 16
    int total = 2 * KT * 256;                // 2 col tiles
    int t = blockIdx.x * 256 + threadIdx.x;
    if (t >= total) return;
    int v = t & 7;
    int lane = (t >> 3) & 31;
    int tile = t >> 8;
    int ktIdx = tile % KT;
    int ct = tile / KT;
    int n = lane & 15, half = lane >> 4;
    int k0 = ktIdx * 32 + half * 16 + 2 * v;
    int col = ct * 16 + n;
    float f0 = 0.f, f1 = 0.f;
    if (col < D1) {
        int hh = col >> 3, o = col & 7;
        f0 = W[((size_t)hh * NFEAT + k0) * NH1 + o];
        f1 = W[((size_t)hh * NFEAT + k0 + 1) * NH1 + o];
    }
    P[t] = (unsigned)f2bf(f0) | ((unsigned)f2bf(f1) << 16);
}

// --------------------------- small utility kernels -------------------------

// neighbor = A_sparse @ x   (512-wide rows, 32 lanes/edge x 16 floats)
__global__ void k_spmm512(const int* __restrict__ idx, const float* __restrict__ vals,
                          const float* __restrict__ X, float* __restrict__ OUT, int E) {
    long long t = (long long)blockIdx.x * 256 + threadIdx.x;
    int e = (int)(t >> 5);
    if (e >= E) return;
    int c = ((int)t & 31) * 16;
    int row = idx[e], col = idx[E + e];
    float v = vals[e];
    const float* xs = X + (size_t)col * NFEAT + c;
    float* op = OUT + (size_t)row * NFEAT + c;
#pragma unroll
    for (int j = 0; j < 16; ++j) atomicAdd(op + j, v * xs[j]);
}

__global__ void k_spmm24(const int* __restrict__ idx, const float* __restrict__ vals,
                         const float* __restrict__ X, float* __restrict__ OUT, int E) {
    int e = blockIdx.x * 256 + threadIdx.x;
    if (e >= E) return;
    int row = idx[e], col = idx[E + e];
    float v = vals[e];
    const float* xs = X + (size_t)col * D1;
    float* op = OUT + (size_t)row * D1;
#pragma unroll
    for (int j = 0; j < D1; ++j) atomicAdd(op + j, v * xs[j]);
}

__global__ void k_sqrt(const float* __restrict__ s, float* __restrict__ o, int n) {
    int i = blockIdx.x * 256 + threadIdx.x;
    if (i < n) o[i] = sqrtf(s[i]);
}

// --------------------- fused layer-1 relation WMMA GEMM --------------------
// accg = x@g1 + nb@g2 ; accb = x@b1 + nb@b2 ; epilogue computes m1, hin(bf16),
// and row-reduced ||m1||^2.
// block = 128 threads (4 waves); grid = (N/16).  Each wave owns 8 col-tiles
// (cols wave*128 .. +127), so x/neighbor are streamed from HBM exactly once.
// A staged via LDS (f32 -> bf16); B fragments loaded directly from packed
// weights (2x global_load_b128 each, L2-resident).

__global__ __launch_bounds__(128)
void k_gemm1_relation(const float* __restrict__ X, const float* __restrict__ NB,
                      const unsigned* __restrict__ Pg1, const unsigned* __restrict__ Pg2,
                      const unsigned* __restrict__ Pb1, const unsigned* __restrict__ Pb2,
                      const float* __restrict__ R1, const int* __restrict__ headp,
                      unsigned short* __restrict__ HINB, float* __restrict__ M1SQ) {
    __shared__ alignas(16) unsigned short atile[16 * 32];        // shared by all 4 waves

    const int wave = threadIdx.x >> 5;
    const int lane = threadIdx.x & 31;
    const int rowbase = blockIdx.x << 4;
    const int ct0 = wave << 3;                                   // first of 8 col-tiles
    const int m = lane & 15, half = lane >> 4;

    v8f zero = {};
    v8f accg[8], accb[8];
#pragma unroll
    for (int i = 0; i < 8; ++i) { accg[i] = zero; accb[i] = zero; }

    for (int phase = 0; phase < 2; ++phase) {
        const float* A = phase ? NB : X;
        const unsigned* Pg = phase ? Pg2 : Pg1;
        const unsigned* Pb = phase ? Pb2 : Pb1;
        for (int ktIdx = 0; ktIdx < NFEAT / 32; ++ktIdx) {
            const int kt = ktIdx << 5;
            __syncthreads();
            { // stage A tile f32 -> bf16 (128 thr x 4 floats = 16x32)
                int t = threadIdx.x << 2;
                int r = t >> 5, c = t & 31;
                const float* ap = A + (size_t)(rowbase + r) * NFEAT + kt + c;
                float4 f = *(const float4*)ap;
                if (kt + 32 < NFEAT) __builtin_prefetch(ap + 32, 0, 1);   // global_prefetch_b8
                unsigned* dp = (unsigned*)(atile + r * 32 + c);
                dp[0] = pack2_bf16(f.x, f.y);
                dp[1] = pack2_bf16(f.z, f.w);
            }
            __syncthreads();
            Frag fa;
#pragma unroll
            for (int v = 0; v < 8; ++v) {
                int ka = half * 16 + (v << 1);   // A 16x32 layout
                fa.u[v] = *(const unsigned*)(atile + m * 32 + ka);
            }
#pragma unroll
            for (int c8 = 0; c8 < 8; ++c8) {
                size_t base = ((size_t)((ct0 + c8) * (NFEAT / 32) + ktIdx) * 32 + lane) << 3;
                Frag fg, fb;
                const uint4* qg = (const uint4*)(Pg + base);
                const uint4* qb = (const uint4*)(Pb + base);
                fg.q[0] = qg[0]; fg.q[1] = qg[1];
                fb.q[0] = qb[0]; fb.q[1] = qb[1];
                accg[c8] = __builtin_amdgcn_wmma_f32_16x16x32_bf16(false, fa.v, false, fg.v, (short)0, accg[c8], false, false);
                accb[c8] = __builtin_amdgcn_wmma_f32_16x16x32_bf16(false, fa.v, false, fb.v, (short)0, accb[c8], false, false);
            }
        }
    }

    // ---- fused _relation epilogue (x/nb tiles are L2-warm: just streamed) ----
    const int head = headp[0];
#pragma unroll
    for (int r = 0; r < 8; ++r) {
        int row = rowbase + r + (half << 3);                // C/D layout: M = r + 8*half
        float s = 0.f;
#pragma unroll
        for (int c8 = 0; c8 < 8; ++c8) {
            int col = ((ct0 + c8) << 4) + m;
            size_t o = (size_t)row * NFEAT + col;
            float xv = X[o], nv = NB[o];
            float gamma = lrelu_f(accg[c8][r]) + 1.f;
            float beta  = lrelu_f(accb[c8][r]);
            float m1 = xv + gamma * R1[col] + beta - nv;
            HINB[o] = f2bf(head ? xv : xv + m1);
            s += m1 * m1;
        }
#pragma unroll
        for (int off = 1; off < 16; off <<= 1) s += __shfl_xor(s, off);
        if (m == 0) atomicAdd(&M1SQ[row], s);
    }
}

// ------------------- single-wave WMMA GEMM, 32 output cols -----------------
// C[M,32] (f32) = A[M,K] (bf16, row-major) @ P (pre-packed B fragments,
// 2 col-tiles).  grid = (M/16); block = 32 (one wave).

__global__ __launch_bounds__(32)
void k_wmma_gemm32(const unsigned short* __restrict__ A, int lda,
                   const unsigned* __restrict__ P,
                   float* __restrict__ C, int ldc, int K) {
    __shared__ alignas(16) unsigned short at[16 * 32];
    const int lane = threadIdx.x;
    const int rowbase = blockIdx.x << 4;
    const int m = lane & 15, half = lane >> 4;
    const int KT = K >> 5;
    v8f zero = {};
    v8f acc[2] = { zero, zero };
    for (int ktIdx = 0; ktIdx < KT; ++ktIdx) {
        const int kt = ktIdx << 5;
        __syncthreads();
        {
            int r = lane >> 1, hh = (lane & 1) << 4;
            const unsigned short* s = A + (size_t)(rowbase + r) * lda + kt + hh;
            uint4* d = (uint4*)(at + r * 32 + hh);
            d[0] = ((const uint4*)s)[0]; d[1] = ((const uint4*)s)[1];
        }
        __syncthreads();
        Frag fa;
#pragma unroll
        for (int v = 0; v < 8; ++v) {
            int ka = half * 16 + (v << 1);
            fa.u[v] = *(const unsigned*)(at + m * 32 + ka);
        }
#pragma unroll
        for (int ct = 0; ct < 2; ++ct) {
            Frag fb;
            size_t base = ((size_t)(ct * KT + ktIdx) * 32 + lane) << 3;
            const uint4* q = (const uint4*)(P + base);
            fb.q[0] = q[0]; fb.q[1] = q[1];
            acc[ct] = __builtin_amdgcn_wmma_f32_16x16x32_bf16(false, fa.v, false, fb.v, (short)0, acc[ct], false, false);
        }
    }
#pragma unroll
    for (int ct = 0; ct < 2; ++ct)
#pragma unroll
        for (int r = 0; r < 8; ++r)
            C[(size_t)(rowbase + r + (half << 3)) * ldc + (ct << 4) + m] = acc[ct][r];
}

// --------------------------- GAT attention stages --------------------------

__global__ void k_gat1_edges(const int* __restrict__ es, int ES,
                             const float* __restrict__ H /*[N][32]*/,
                             const float* __restrict__ a /*[3][16]*/,
                             float* __restrict__ rowsum /*[3][N]*/,
                             float* __restrict__ hp /*[N][24]*/, int N) {
    int e = blockIdx.x * 256 + threadIdx.x;
    if (e >= ES) return;
    int s = es[e], d = es[ES + e];
    const float* hs = H + (size_t)s * 32;
    const float* hd = H + (size_t)d * 32;
#pragma unroll
    for (int hh = 0; hh < NHEADS; ++hh) {
        float z = 0.f;
#pragma unroll
        for (int o = 0; o < NH1; ++o)
            z = fmaf(a[hh * 16 + o], hs[hh * 8 + o], fmaf(a[hh * 16 + 8 + o], hd[hh * 8 + o], z));
        float ee = expf(-lrelu_f(z));
        atomicAdd(rowsum + (size_t)hh * N + s, ee);
#pragma unroll
        for (int o = 0; o < NH1; ++o)
            atomicAdd(hp + (size_t)s * D1 + hh * 8 + o, ee * hd[hh * 8 + o]);
    }
}

// x1 = elu(elu(hp / rowsum))
__global__ void k_x1(const float* __restrict__ hp, const float* __restrict__ rowsum,
                     float* __restrict__ x1, int N) {
    int t = blockIdx.x * 256 + threadIdx.x;
    if (t >= N * D1) return;
    int i = t / D1, c = t - i * D1, hh = c >> 3;
    float v = hp[t] / rowsum[(size_t)hh * N + i];
    x1[t] = elu_f(elu_f(v));
}

// layer-2 relation (24x24, weights in LDS); writes hin2 and ||m2||
__global__ __launch_bounds__(256)
void k_relation2(const float* __restrict__ X1, const float* __restrict__ NB2,
                 const float* __restrict__ G1, const float* __restrict__ G2,
                 const float* __restrict__ B1, const float* __restrict__ B2,
                 const float* __restrict__ R2, const int* __restrict__ headp,
                 float* __restrict__ HIN2, float* __restrict__ M2N, int N) {
    __shared__ float g1s[D1 * D1], g2s[D1 * D1], b1s[D1 * D1], b2s[D1 * D1], r2s[D1];
    for (int t = threadIdx.x; t < D1 * D1; t += 256) {
        g1s[t] = G1[t]; g2s[t] = G2[t]; b1s[t] = B1[t]; b2s[t] = B2[t];
    }
    if (threadIdx.x < D1) r2s[threadIdx.x] = R2[threadIdx.x];
    __syncthreads();
    int i = blockIdx.x * 256 + threadIdx.x;
    if (i >= N) return;
    float xr[D1], nr[D1];
#pragma unroll
    for (int k = 0; k < D1; ++k) { xr[k] = X1[(size_t)i * D1 + k]; nr[k] = NB2[(size_t)i * D1 + k]; }
    int head = headp[0];
    float s = 0.f;
    for (int j = 0; j < D1; ++j) {
        float G = 0.f, B = 0.f;
#pragma unroll
        for (int k = 0; k < D1; ++k) {
            G = fmaf(xr[k], g1s[k * D1 + j], fmaf(nr[k], g2s[k * D1 + j], G));
            B = fmaf(xr[k], b1s[k * D1 + j], fmaf(nr[k], b2s[k * D1 + j], B));
        }
        float m2 = xr[j] + (lrelu_f(G) + 1.f) * r2s[j] + lrelu_f(B) - nr[j];
        s += m2 * m2;
        HIN2[(size_t)i * D1 + j] = head ? xr[j] : xr[j] + m2;
    }
    M2N[i] = sqrtf(s);
}

// hg = hin2 @ gat2_W  ([24][32] weights in LDS)
__global__ __launch_bounds__(256)
void k_hgat2(const float* __restrict__ HIN2, const float* __restrict__ W,
             float* __restrict__ HG, int N) {
    __shared__ float ws[D1 * NCLASS];
    for (int t = threadIdx.x; t < D1 * NCLASS; t += 256) ws[t] = W[t];
    __syncthreads();
    int i = blockIdx.x * 256 + threadIdx.x;
    if (i >= N) return;
    float hr[D1];
#pragma unroll
    for (int k = 0; k < D1; ++k) hr[k] = HIN2[(size_t)i * D1 + k];
    for (int c = 0; c < NCLASS; ++c) {
        float a = 0.f;
#pragma unroll
        for (int k = 0; k < D1; ++k) a = fmaf(hr[k], ws[k * NCLASS + c], a);
        HG[(size_t)i * NCLASS + c] = a;
    }
}

__global__ void k_gat2_edges(const int* __restrict__ es, int ES,
                             const float* __restrict__ HG /*[N][32]*/,
                             const float* __restrict__ a /*[64]*/,
                             float* __restrict__ rowsum /*[N]*/,
                             float* __restrict__ hp /*[N][32]*/) {
    int e = blockIdx.x * 256 + threadIdx.x;
    if (e >= ES) return;
    int s = es[e], d = es[ES + e];
    const float* hs = HG + (size_t)s * NCLASS;
    const float* hd = HG + (size_t)d * NCLASS;
    float z = 0.f;
#pragma unroll
    for (int c = 0; c < NCLASS; ++c)
        z = fmaf(a[c], hs[c], fmaf(a[NCLASS + c], hd[c], z));
    float ee = expf(-lrelu_f(z));
    atomicAdd(rowsum + s, ee);
#pragma unroll
    for (int c = 0; c < NCLASS; ++c)
        atomicAdd(hp + (size_t)s * NCLASS + c, ee * hd[c]);
}

__global__ void k_x2(const float* __restrict__ hp, const float* __restrict__ rowsum,
                     float* __restrict__ out, int N) {
    int t = blockIdx.x * 256 + threadIdx.x;
    if (t >= N * NCLASS) return;
    int i = t / NCLASS;
    out[t] = elu_f(hp[t] / rowsum[i]);
}

// -------------------------------- launcher ---------------------------------

extern "C" void kernel_launch(void* const* d_in, const int* in_sizes, int n_in,
                              void* d_out, int out_size, void* d_ws, size_t ws_size,
                              hipStream_t stream) {
    if (n_in < 19) return;
    const float* x       = (const float*)d_in[0];
    const float* vals    = (const float*)d_in[1];
    const float* w_g1    = (const float*)d_in[2];
    const float* w_g2    = (const float*)d_in[3];
    const float* w_b1    = (const float*)d_in[4];
    const float* w_b2    = (const float*)d_in[5];
    const float* r1_r    = (const float*)d_in[6];
    const float* gat1_W  = (const float*)d_in[7];
    const float* gat1_a  = (const float*)d_in[8];
    const float* w2_g1   = (const float*)d_in[9];
    const float* w2_g2   = (const float*)d_in[10];
    const float* w2_b1   = (const float*)d_in[11];
    const float* w2_b2   = (const float*)d_in[12];
    const float* r2_r    = (const float*)d_in[13];
    const float* gat2_W  = (const float*)d_in[14];
    const float* gat2_a  = (const float*)d_in[15];
    const int*   adj_idx = (const int*)d_in[16];
    const int*   es      = (const int*)d_in[17];
    const int*   headp   = (const int*)d_in[18];

    const int N  = in_sizes[0] / NFEAT;       // 100000 (multiple of 16)
    const int E  = in_sizes[1];               // 1600000
    const int ES = in_sizes[17] / 2;          // E + N

    float* out     = (float*)d_out;
    float* out_x2  = out;                       // [N*32]
    float* out_m1n = out + (size_t)N * NCLASS;  // [N]
    float* out_m2n = out_m1n + N;               // [N]

    // ---- workspace carve-out (256B aligned) ----
    char* wsp = (char*)d_ws;
    size_t off = 0;
    auto carve = [&](size_t bytes) { void* p = wsp + off; off += (bytes + 255) & ~(size_t)255; return p; };
    const size_t packW = (size_t)(NFEAT / 16) * (NFEAT / 32) * 256 * 4;   // 512 KB
    float*          nbr  = (float*)carve((size_t)N * NFEAT * 4);          // region0 (reused later)
    unsigned short* hinb = (unsigned short*)carve((size_t)N * NFEAT * 2);
    unsigned*       Pg1  = (unsigned*)carve(packW);
    unsigned*       Pg2  = (unsigned*)carve(packW);
    unsigned*       Pb1  = (unsigned*)carve(packW);
    unsigned*       Pb2  = (unsigned*)carve(packW);
    unsigned*       Pw1  = (unsigned*)carve((size_t)2 * (NFEAT / 32) * 256 * 4); // 32 KB
    float*          m1sq = (float*)carve((size_t)N * 4);
    float*          x1   = (float*)carve((size_t)N * D1 * 4);
    float*          nb2  = (float*)carve((size_t)N * D1 * 4);
    (void)ws_size;
    // region0 reuse (neighbor is dead after the fused layer-1 GEMM)
    char* r0 = (char*)nbr;
    float* h    = (float*)r0;                 r0 += (size_t)N * 32 * 4;
    float* hp1  = (float*)r0;                 r0 += (size_t)N * D1 * 4;
    float* rs1  = (float*)r0;                 r0 += (size_t)NHEADS * N * 4;
    float* hg   = (float*)r0;                 r0 += (size_t)N * NCLASS * 4;
    float* hp2  = (float*)r0;                 r0 += (size_t)N * NCLASS * 4;
    float* rs2  = (float*)r0;                 r0 += (size_t)N * 4;
    float* hin2 = (float*)r0;                 r0 += (size_t)N * D1 * 4;

    // 1) weight packing into WMMA fragment layout (one-time, L2-resident)
    const int nPack = (NFEAT / 16) * (NFEAT / 32) * 256;     // 131072 u32 per matrix
    k_pack_wfrag<<<(nPack + 255) / 256, 256, 0, stream>>>(w_g1, Pg1, NFEAT, NFEAT, NFEAT);
    k_pack_wfrag<<<(nPack + 255) / 256, 256, 0, stream>>>(w_g2, Pg2, NFEAT, NFEAT, NFEAT);
    k_pack_wfrag<<<(nPack + 255) / 256, 256, 0, stream>>>(w_b1, Pb1, NFEAT, NFEAT, NFEAT);
    k_pack_wfrag<<<(nPack + 255) / 256, 256, 0, stream>>>(w_b2, Pb2, NFEAT, NFEAT, NFEAT);
    k_pack_w1frag<<<(2 * (NFEAT / 32) * 256 + 255) / 256, 256, 0, stream>>>(gat1_W, Pw1);

    // 2) neighbor = spmm(adj, x)
    hipMemsetAsync(nbr, 0, (size_t)N * NFEAT * 4, stream);
    {
        long long tt = (long long)E * 32;
        k_spmm512<<<(unsigned)((tt + 255) / 256), 256, 0, stream>>>(adj_idx, vals, x, nbr, E);
    }

    // 3) fused relation GEMM (WMMA, single pass over x/nb) -> hinb, m1sq
    hipMemsetAsync(m1sq, 0, (size_t)N * 4, stream);
    k_gemm1_relation<<<N / 16, 128, 0, stream>>>(x, nbr, Pg1, Pg2, Pb1, Pb2,
                                                 r1_r, headp, hinb, m1sq);
    k_sqrt<<<(N + 255) / 256, 256, 0, stream>>>(m1sq, out_m1n, N);

    // 4) h = hin @ gat1_W (padded to 32 cols) via single-wave WMMA GEMM
    k_wmma_gemm32<<<N / 16, 32, 0, stream>>>(hinb, NFEAT, Pw1, h, 32, NFEAT);

    // 5) layer-1 GAT attention
    hipMemsetAsync(hp1, 0, (size_t)N * D1 * 4, stream);
    hipMemsetAsync(rs1, 0, (size_t)NHEADS * N * 4, stream);
    k_gat1_edges<<<(ES + 255) / 256, 256, 0, stream>>>(es, ES, h, gat1_a, rs1, hp1, N);
    k_x1<<<(N * D1 + 255) / 256, 256, 0, stream>>>(hp1, rs1, x1, N);

    // 6) layer-2: spmm, relation, projection
    hipMemsetAsync(nb2, 0, (size_t)N * D1 * 4, stream);
    k_spmm24<<<(E + 255) / 256, 256, 0, stream>>>(adj_idx, vals, x1, nb2, E);
    k_relation2<<<(N + 255) / 256, 256, 0, stream>>>(x1, nb2, w2_g1, w2_g2, w2_b1, w2_b2,
                                                     r2_r, headp, hin2, out_m2n, N);
    k_hgat2<<<(N + 255) / 256, 256, 0, stream>>>(hin2, gat2_W, hg, N);

    // 7) layer-2 GAT attention -> x2
    hipMemsetAsync(hp2, 0, (size_t)N * NCLASS * 4, stream);
    hipMemsetAsync(rs2, 0, (size_t)N * 4, stream);
    k_gat2_edges<<<(ES + 255) / 256, 256, 0, stream>>>(es, ES, hg, gat2_a, rs2, hp2);
    k_x2<<<(N * NCLASS + 255) / 256, 256, 0, stream>>>(hp2, rs2, out_x2, N);
    (void)out_size;
}
